// end_to_end_model_86019605004745
// MI455X (gfx1250) — compile-verified, matrix operand order checked by reading, not run on hardware
//
#include <hip/hip_runtime.h>
#include <cstdint>

// ---------------------------------------------------------------------------
// CDNA5 (gfx1250) end-to-end model.  All GEMM-shaped work runs on
// v_wmma_f32_16x16x32_f16 (f16 operands, f32 accumulate), wave32.
// Operands are pre-converted to f16 once (weights at launch, activations at
// their producer pass) so WMMA inner loops are pure b128 loads + wmma.
// Big GEMMs stage 64x32 operand slabs into LDS via the gfx1250 async
// global->LDS copy path (global_load_async_to_lds_b128 + s_wait_asynccnt).
// ---------------------------------------------------------------------------

typedef __attribute__((ext_vector_type(16))) _Float16 v16h;
typedef __attribute__((ext_vector_type(8)))  _Float16 v8h;
typedef __attribute__((ext_vector_type(8)))  float    v8f;
typedef int v4i_ __attribute__((vector_size(16)));   // matches builtin param type

#define DEVFN static __device__ __forceinline__
#define WMMA_F16(a, b, c) \
  __builtin_amdgcn_wmma_f32_16x16x32_f16(false, (a), false, (b), (short)0, (c), false, false)

#if defined(__has_builtin)
#if __has_builtin(__builtin_amdgcn_global_load_async_to_lds_b128)
#define HAVE_ASYNC_COPY 1
#endif
#endif

// ---------------------------------------------------------------------------
// Fragment helpers (CDNA5 16-bit A/B layout, 16x16x32):
// lane l: row/col = l&15 ; K base kb = 8*(l>>4)
// elements 0..7  -> K = kb+0..kb+7    ; elements 8..15 -> K = kb+16..kb+23
// Both chunks are 8 contiguous halves (16B) -> two b128 loads per fragment.
// ---------------------------------------------------------------------------
DEVFN v16h frag_combine(v8h lo, v8h hi) {
  return __builtin_shufflevector(lo, hi, 0, 1, 2, 3, 4, 5, 6, 7, 8, 9, 10, 11, 12, 13, 14, 15);
}

DEVFN v16h frag_zero() {
  v16h f;
#pragma unroll
  for (int i = 0; i < 16; ++i) f[i] = (_Float16)0.0f;
  return f;
}

// Row-major f16 matrix, fixed row, K-slice at k0.  Requires (ld % 8 == 0).
DEVFN v16h load_frag_h(const _Float16* __restrict__ p, long ld, int row, int k0, int lane) {
  int kb = (lane >> 4) << 3;
  const _Float16* b = p + (long)row * ld + k0 + kb;
  v8h lo = *(const v8h*)b;
  v8h hi = *(const v8h*)(b + 16);
  return frag_combine(lo, hi);
}

// Fragment from an LDS slab laid out [64 rows][32 halves].
DEVFN v16h load_frag_lds(const _Float16* p, int row, int lane) {
  int kb = (lane >> 4) << 3;
  const _Float16* b = p + row * 32 + kb;
  v8h lo = *(const v8h*)b;
  v8h hi = *(const v8h*)(b + 16);
  return frag_combine(lo, hi);
}

// 16-byte global -> LDS copy; async (ASYNCcnt) when available.
DEVFN void copy16_g2l(const _Float16* __restrict__ g, _Float16* l) {
#ifdef HAVE_ASYNC_COPY
  __attribute__((address_space(1))) v4i_* gp =
      (__attribute__((address_space(1))) v4i_*)(uintptr_t)(const void*)g;
  __attribute__((address_space(3))) v4i_* lp =
      (__attribute__((address_space(3))) v4i_*)(uint32_t)(uintptr_t)(void*)l;
  __builtin_amdgcn_global_load_async_to_lds_b128(gp, lp, 0, 0);
#else
  *(v8h*)l = *(const v8h*)g;
#endif
}

DEVFN void wait_async_copies() {
#ifdef HAVE_ASYNC_COPY
  asm volatile("s_wait_asynccnt 0x0" ::: "memory");
#endif
}

// C/D tile store: VGPR j <-> M = tm + j + 8*(lane>>4), lane&15 <-> N.
template <typename OutT>
DEVFN void store_tile(const v8f& acc, OutT* __restrict__ O, long ldo, int tm, int tn,
                      int M, int N, const float* __restrict__ bp, int relu, int lane) {
  int n = tn + (lane & 15);
  if (n >= N) return;
  float bv = bp ? bp[n] : 0.0f;
#pragma unroll
  for (int j = 0; j < 8; ++j) {
    int m = tm + j + ((lane >> 4) << 3);
    if (m < M) {
      float v = acc[j] + bv;
      if (relu) v = fmaxf(v, 0.0f);
      O[(long)m * ldo + n] = (OutT)v;
    }
  }
}

// ---------------------------------------------------------------------------
// Direct GEMM (f16 operands): O = act(A * W^T + bias).  2x2 tiles per wave.
// A [M x K] (lda), W [N x K] (ldw); K%32==0, M%32==0; N tail guarded.
// ---------------------------------------------------------------------------
template <typename OutT>
__global__ void gemm_bt_h_kernel(const _Float16* __restrict__ A, long lda, long strideA,
                                 const _Float16* __restrict__ W, long ldw, long strideW,
                                 const float* __restrict__ bias, long strideBias,
                                 OutT* __restrict__ O, long ldo, long strideO,
                                 int M, int N, int K, int relu) {
  int lane = threadIdx.x & 31;
  int wave = threadIdx.x >> 5;
  int z = blockIdx.z;
  A += (long)z * strideA;
  W += (long)z * strideW;
  O += (long)z * strideO;
  const float* bp = bias ? bias + (long)z * strideBias : nullptr;

  int tm = blockIdx.y * 64 + (wave >> 1) * 32;
  int tn = blockIdx.x * 64 + (wave & 1) * 32;
  if (tm >= M) return;  // uniform per wave

  int r0 = tm + (lane & 15), r1 = r0 + 16;
  int c0 = tn + (lane & 15), c1 = c0 + 16;
  bool ok0 = c0 < N, ok1 = c1 < N;

  v8f a00 = {}, a01 = {}, a10 = {}, a11 = {};
  for (int k0 = 0; k0 < K; k0 += 32) {
    __builtin_prefetch(A + (long)r0 * lda + k0 + 256, 0, 1);  // global_prefetch_b8
    v16h fa0 = load_frag_h(A, lda, r0, k0, lane);
    v16h fa1 = load_frag_h(A, lda, r1, k0, lane);
    v16h fb0 = ok0 ? load_frag_h(W, ldw, c0, k0, lane) : frag_zero();
    v16h fb1 = ok1 ? load_frag_h(W, ldw, c1, k0, lane) : frag_zero();
    a00 = WMMA_F16(fa0, fb0, a00);
    a01 = WMMA_F16(fa0, fb1, a01);
    a10 = WMMA_F16(fa1, fb0, a10);
    a11 = WMMA_F16(fa1, fb1, a11);
  }
  store_tile(a00, O, ldo, tm,      tn,      M, N, bp, relu, lane);
  store_tile(a01, O, ldo, tm,      tn + 16, M, N, bp, relu, lane);
  store_tile(a10, O, ldo, tm + 16, tn,      M, N, bp, relu, lane);
  store_tile(a11, O, ldo, tm + 16, tn + 16, M, N, bp, relu, lane);
}

// ---------------------------------------------------------------------------
// LDS-staged GEMM: block = 128 threads (4 waves) computing a 64x64 tile.
// Per K-step both 64x32 slabs are staged to LDS (async path when available),
// then each wave runs 4 WMMAs from ds_load_b128 fragments.
// Requires M%64==0, N%64==0, K%32==0.
// ---------------------------------------------------------------------------
template <typename OutT>
__global__ void gemm_bt_lds_kernel(const _Float16* __restrict__ A, long lda, long strideA,
                                   const _Float16* __restrict__ W, long ldw, long strideW,
                                   const float* __restrict__ bias, long strideBias,
                                   OutT* __restrict__ O, long ldo, long strideO,
                                   int M, int N, int K, int relu) {
  __shared__ __align__(16) _Float16 Ash[64 * 32];
  __shared__ __align__(16) _Float16 Wsh[64 * 32];

  int tid = threadIdx.x;
  int lane = tid & 31;
  int wave = tid >> 5;
  int z = blockIdx.z;
  const _Float16* Ab = A + (long)z * strideA + (long)blockIdx.y * 64 * lda;
  const _Float16* Wb = W + (long)z * strideW + (long)blockIdx.x * 64 * ldw;
  O += (long)z * strideO;
  const float* bp = bias ? bias + (long)z * strideBias : nullptr;

  int tm = blockIdx.y * 64 + (wave >> 1) * 32;
  int tn = blockIdx.x * 64 + (wave & 1) * 32;
  int rl = (wave >> 1) * 32 + (lane & 15);  // block-local rows/cols
  int cl = (wave & 1) * 32 + (lane & 15);

  v8f a00 = {}, a01 = {}, a10 = {}, a11 = {};
  for (int k0 = 0; k0 < K; k0 += 32) {
    // Stage A and W slabs: 256 16B-units each, 2 per thread per slab.
#pragma unroll
    for (int u0 = 0; u0 < 2; ++u0) {
      int u = tid + u0 * 128;
      int row = u >> 2, seg = (u & 3) * 8;
      copy16_g2l(Ab + (long)row * lda + k0 + seg, &Ash[row * 32 + seg]);
      copy16_g2l(Wb + (long)row * ldw + k0 + seg, &Wsh[row * 32 + seg]);
    }
    wait_async_copies();
    __syncthreads();

    v16h fa0 = load_frag_lds(Ash, rl, lane);
    v16h fa1 = load_frag_lds(Ash, rl + 16, lane);
    v16h fb0 = load_frag_lds(Wsh, cl, lane);
    v16h fb1 = load_frag_lds(Wsh, cl + 16, lane);
    a00 = WMMA_F16(fa0, fb0, a00);
    a01 = WMMA_F16(fa0, fb1, a01);
    a10 = WMMA_F16(fa1, fb0, a10);
    a11 = WMMA_F16(fa1, fb1, a11);
    __syncthreads();
  }
  store_tile(a00, O, ldo, tm,      tn,      M, N, bp, relu, lane);
  store_tile(a01, O, ldo, tm,      tn + 16, M, N, bp, relu, lane);
  store_tile(a10, O, ldo, tm + 16, tn,      M, N, bp, relu, lane);
  store_tile(a11, O, ldo, tm + 16, tn + 16, M, N, bp, relu, lane);
}

// ---------------------------------------------------------------------------
// Chebyshev recursion step: slot[kcur] = alpha*(Lr @ slot[kprev]) + beta*slot[kprev2]
// X element (b,v,f,k) at Xk[(b*V+v)*KF + f*Kc + k], all f16, f32 accumulate.
// ---------------------------------------------------------------------------
__global__ void cheby_step_kernel(const _Float16* __restrict__ Lr, int V,
                                  _Float16* __restrict__ Xk, int KF, int Kc,
                                  int kprev, int kprev2, int kcur,
                                  float alpha, float beta) {
  int lane = threadIdx.x & 31;
  int wave = threadIdx.x >> 5;
  int b = blockIdx.z;
  int tm = wave * 16;
  int tn = blockIdx.x * 16;
  if (tm >= V) return;

  _Float16* base = Xk + (long)b * V * KF;
  const _Float16* Bp = base + kprev;  // (u,f) at Bp[u*KF + f*Kc]
  int rowA = tm + (lane & 15);
  int col = tn + (lane & 15);
  int kb = (lane >> 4) << 3;

  v8f acc = {};
  for (int k0 = 0; k0 < V; k0 += 32) {
    v16h a = load_frag_h(Lr, V, rowA, k0, lane);
    v16h bf;
#pragma unroll
    for (int i = 0; i < 16; ++i) {
      int k = k0 + kb + (i < 8 ? i : i + 8);
      bf[i] = Bp[(long)k * KF + (long)col * Kc];
    }
    acc = WMMA_F16(a, bf, acc);
  }
#pragma unroll
  for (int j = 0; j < 8; ++j) {
    int v = tm + j + ((lane >> 4) << 3);
    int f = tn + (lane & 15);
    float val = alpha * acc[j];
    if (beta != 0.0f) val += beta * (float)base[(long)v * KF + (long)f * Kc + kprev2];
    base[(long)v * KF + (long)f * Kc + kcur] = (_Float16)val;
  }
}

// ---------------------------------------------------------------------------
// conv1 (VALU, ~1 GFLOP): y1[(c*32+co)][b*249+l] = relu(conv + bias), f32 out
// ---------------------------------------------------------------------------
__global__ void conv1_kernel(const float* __restrict__ inp, const float* __restrict__ w1,
                             const float* __restrict__ b1, float* __restrict__ y1) {
  int g = blockIdx.y;  // c*32+co
  int c = g >> 5;
  int idx = blockIdx.x * blockDim.x + threadIdx.x;  // b*249 + l
  if (idx >= 64 * 249) return;
  int b = idx / 249, l = idx % 249;
  const float* xin = inp + ((long)b * 64 + c) * 512 + 2 * l;
  const float* wr = w1 + (long)g * 16;
  float s = b1[g];
#pragma unroll
  for (int t = 0; t < 16; ++t) s += wr[t] * xin[t];
  y1[(long)g * (64 * 249) + idx] = fmaxf(s, 0.0f);
}

// ---------------------------------------------------------------------------
// conv2 implicit GEMM (WMMA): M=co(64), N=b*55+l(3520), K=ci*16+t(512)
// o1h [c][b][32][124] f16, w2h [c][64][512] f16 -> y2 [c][64][3520] f32 (relu)
// ---------------------------------------------------------------------------
__global__ void conv2_wmma_kernel(const _Float16* __restrict__ o1h, const _Float16* __restrict__ w2h,
                                  const float* __restrict__ b2, float* __restrict__ y2) {
  const int L2 = 55, Lp1 = 124, N2 = 64 * 55;
  int lane = threadIdx.x & 31;
  int wave = threadIdx.x >> 5;
  int c = blockIdx.z;
  int tm = wave * 16;
  int tn = blockIdx.x * 16;
  int rowA = tm + (lane & 15);
  int n = tn + (lane & 15);
  int bb = n / L2, ll = n % L2;
  const _Float16* Bsrc = o1h + (((long)c * 64 + bb) * 32) * Lp1 + 2 * ll;
  const _Float16* Asrc = w2h + (long)c * 64 * 512;
  int kb = (lane >> 4) << 3;

  v8f acc = {};
  for (int k0 = 0; k0 < 512; k0 += 32) {
    v16h a = load_frag_h(Asrc, 512, rowA, k0, lane);
    v16h bf;
#pragma unroll
    for (int i = 0; i < 16; ++i) {
      int k = k0 + kb + (i < 8 ? i : i + 8);
      int ci = k >> 4, t = k & 15;
      bf[i] = Bsrc[(long)ci * Lp1 + t];
    }
    acc = WMMA_F16(a, bf, acc);
  }
#pragma unroll
  for (int j = 0; j < 8; ++j) {
    int co = tm + j + ((lane >> 4) << 3);
    float v = acc[j] + b2[c * 64 + co];
    y2[((long)c * 64 + co) * N2 + n] = fmaxf(v, 0.0f);
  }
}

// --- mean/var over contiguous groups (f32 input) ---------------------------
__global__ void group_stats_kernel(const float* __restrict__ x, long groupStride, int n,
                                   float* __restrict__ mean, float* __restrict__ var) {
  __shared__ float s1[256], s2[256];
  int g = blockIdx.x;
  const float* p = x + (long)g * groupStride;
  float a = 0.0f, q = 0.0f;
  for (int i = threadIdx.x; i < n; i += blockDim.x) {
    float v = p[i];
    a += v; q += v * v;
  }
  s1[threadIdx.x] = a; s2[threadIdx.x] = q;
  __syncthreads();
  for (int o = blockDim.x >> 1; o > 0; o >>= 1) {
    if ((int)threadIdx.x < o) { s1[threadIdx.x] += s1[threadIdx.x + o]; s2[threadIdx.x] += s2[threadIdx.x + o]; }
    __syncthreads();
  }
  if (threadIdx.x == 0) {
    float m = s1[0] / n;
    mean[g] = m;
    var[g] = s2[0] / n - m * m;
  }
}

// Per-node stats over (B,F) for graph BN (relu applied); x f32 [b][v][f].
__global__ void node_stats_kernel(const float* __restrict__ x, int V, int F, int NB,
                                  float* __restrict__ mean, float* __restrict__ var) {
  __shared__ float s1[256], s2[256];
  int g = blockIdx.x;
  float a = 0.0f, q = 0.0f;
  int n = NB * F;
  for (int i = threadIdx.x; i < n; i += blockDim.x) {
    int b = i / F, f = i % F;
    float v = fmaxf(x[((long)b * V + g) * F + f], 0.0f);
    a += v; q += v * v;
  }
  s1[threadIdx.x] = a; s2[threadIdx.x] = q;
  __syncthreads();
  for (int o = blockDim.x >> 1; o > 0; o >>= 1) {
    if ((int)threadIdx.x < o) { s1[threadIdx.x] += s1[threadIdx.x + o]; s2[threadIdx.x] += s2[threadIdx.x + o]; }
    __syncthreads();
  }
  if (threadIdx.x == 0) {
    float m = s1[0] / n;
    mean[g] = m;
    var[g] = s2[0] / n - m * m;
  }
}

// BN1 apply + maxpool2 -> f16 o1h [((c*64+b)*32+co)*124+lp]
__global__ void bn1_pool_kernel(const float* __restrict__ y1, const float* __restrict__ mean,
                                const float* __restrict__ var, const float* __restrict__ g1,
                                const float* __restrict__ be1, _Float16* __restrict__ o1h) {
  long idx = (long)blockIdx.x * blockDim.x + threadIdx.x;
  const long total = 64L * 32 * 64 * 124;
  if (idx >= total) return;
  int lp = (int)(idx % 124); long r = idx / 124;
  int b = (int)(r % 64); r /= 64;
  int co = (int)(r % 32); int c = (int)(r / 32);
  int grp = c * 32 + co;
  float m = mean[grp], iv = rsqrtf(var[grp] + 1e-5f);
  float sc = g1[grp] * iv, sh = be1[grp] - m * sc;
  const float* p = y1 + (long)grp * (64 * 249) + (long)b * 249 + 2 * lp;
  float v0 = p[0] * sc + sh, v1 = p[1] * sc + sh;
  o1h[(((long)c * 64 + b) * 32 + co) * 124 + lp] = (_Float16)fmaxf(v0, v1);
}

// BN2 apply + maxpool2 -> f16 o2h [(c*64+b)*1728 + co*27 + lp]
__global__ void bn2_pool_kernel(const float* __restrict__ y2, const float* __restrict__ mean,
                                const float* __restrict__ var, const float* __restrict__ g2,
                                const float* __restrict__ be2, _Float16* __restrict__ o2h) {
  long idx = (long)blockIdx.x * blockDim.x + threadIdx.x;
  const long total = 64L * 64 * 64 * 27;
  if (idx >= total) return;
  int lp = (int)(idx % 27); long r = idx / 27;
  int b = (int)(r % 64); r /= 64;
  int co = (int)(r % 64); int c = (int)(r / 64);
  int grp = c * 64 + co;
  float m = mean[grp], iv = rsqrtf(var[grp] + 1e-5f);
  float sc = g2[grp] * iv, sh = be2[grp] - m * sc;
  const float* p = y2 + (long)grp * 3520 + (long)b * 55 + 2 * lp;
  float v0 = p[0] * sc + sh, v1 = p[1] * sc + sh;
  o2h[((long)c * 64 + b) * 1728 + co * 27 + lp] = (_Float16)fmaxf(v0, v1);
}

// x0 = cout[perm[d], b, :] -> Xk1h slot k=0 (f16, stride-5 feature layout)
__global__ void scatter_x0_kernel(const _Float16* __restrict__ couth, const int* __restrict__ perm,
                                  _Float16* __restrict__ Xk1h) {
  int idx = blockIdx.x * blockDim.x + threadIdx.x;
  if (idx >= 64 * 64 * 512) return;
  int f = idx % 512; int r = idx / 512;
  int d = r % 64; int b = r / 64;
  int src = perm[d];
  Xk1h[((long)b * 64 + d) * 2560 + (long)f * 5] = couth[((long)src * 64 + b) * 512 + f];
}

// Lr = 2*L/lmax - I  (written as f16 WMMA operand)
__global__ void build_lr_kernel(const float* __restrict__ L0, const float* __restrict__ L1,
                                const int* __restrict__ lmax0, const int* __restrict__ lmax1,
                                _Float16* __restrict__ Lr0h, _Float16* __restrict__ Lr1h) {
  int i = blockIdx.x * blockDim.x + threadIdx.x;
  if (i < 64 * 64) {
    int r = i >> 6, cc = i & 63;
    Lr0h[i] = (_Float16)(2.0f * L0[i] / (float)lmax0[0] - (r == cc ? 1.0f : 0.0f));
  }
  if (i < 32 * 32) {
    int r = i >> 5, cc = i & 31;
    Lr1h[i] = (_Float16)(2.0f * L1[i] / (float)lmax1[0] - (r == cc ? 1.0f : 0.0f));
  }
}

// relu + graph-BN apply + graph maxpool (p=2 over nodes) -> strided f16 out
__global__ void gbn_pool_scatter_kernel(const float* __restrict__ x, int V, int F,
                                        const float* __restrict__ mean, const float* __restrict__ var,
                                        float eps, const float* __restrict__ gg,
                                        const float* __restrict__ gb,
                                        _Float16* __restrict__ out, int KF, int Kc) {
  int idx = blockIdx.x * blockDim.x + threadIdx.x;
  int half = V >> 1;
  int total = 64 * half * F;
  if (idx >= total) return;
  int f = idx % F; int r = idx / F;
  int vp = r % half; int b = r / half;
  int v0 = 2 * vp, v1 = 2 * vp + 1;
  float s0 = gg[v0] * rsqrtf(var[v0] + eps);
  float s1 = gg[v1] * rsqrtf(var[v1] + eps);
  float a0 = fmaxf(x[((long)b * V + v0) * F + f], 0.0f);
  float a1 = fmaxf(x[((long)b * V + v1) * F + f], 0.0f);
  float n0 = (a0 - mean[v0]) * s0 + gb[v0];
  float n1 = (a1 - mean[v1]) * s1 + gb[v1];
  out[((long)b * half + vp) * KF + (long)f * Kc] = (_Float16)fmaxf(n0, n1);
}

// one-shot f32 -> f16 conversion (weights etc.)
__global__ void f32_to_f16_kernel(const float* __restrict__ src, _Float16* __restrict__ dst, long n) {
  long i = (long)blockIdx.x * blockDim.x + threadIdx.x;
  if (i < n) dst[i] = (_Float16)src[i];
}

// ---------------------------------------------------------------------------
// Host-side orchestration
// ---------------------------------------------------------------------------
static inline int cdiv(long a, long b) { return (int)((a + b - 1) / b); }

extern "C" void kernel_launch(void* const* d_in, const int* in_sizes, int n_in,
                              void* d_out, int out_size, void* d_ws, size_t ws_size,
                              hipStream_t stream) {
  (void)in_sizes; (void)n_in; (void)out_size; (void)ws_size;

  const float* inp     = (const float*)d_in[0];
  const float* L0      = (const float*)d_in[2];
  const float* L1      = (const float*)d_in[3];
  const int*   lmax0   = (const int*)d_in[4];
  const int*   lmax1   = (const int*)d_in[5];
  const int*   perm    = (const int*)d_in[6];
  const float* conv1_w = (const float*)d_in[8];
  const float* conv1_b = (const float*)d_in[9];
  const float* bn1_g   = (const float*)d_in[10];
  const float* bn1_b   = (const float*)d_in[11];
  const float* conv2_w = (const float*)d_in[12];
  const float* conv2_b = (const float*)d_in[13];
  const float* bn2_g   = (const float*)d_in[14];
  const float* bn2_b   = (const float*)d_in[15];
  const float* lin1_w  = (const float*)d_in[16];
  const float* lin1_b  = (const float*)d_in[17];
  const float* cl1_w   = (const float*)d_in[18];
  const float* cl1_b   = (const float*)d_in[19];
  const float* gbn1_g  = (const float*)d_in[20];
  const float* gbn1_b  = (const float*)d_in[21];
  const float* cl2_w   = (const float*)d_in[22];
  const float* cl2_b   = (const float*)d_in[23];
  const float* gbn2_g  = (const float*)d_in[24];
  const float* gbn2_b  = (const float*)d_in[25];
  const float* fc1_w   = (const float*)d_in[26];
  const float* fc1_b   = (const float*)d_in[27];
  const float* fc2_w   = (const float*)d_in[28];
  const float* fc2_b   = (const float*)d_in[29];
  float* out = (float*)d_out;

  // ---- workspace layout --------------------------------------------------
  const long SZ_A = 64L * 32 * 64 * 249;  // f32 region (y1, then y2, then outs)
  float* regA = (float*)d_ws;
  _Float16* regH = (_Float16*)(regA + SZ_A);

  float* y1   = regA;                     // [2048][15936] f32
  float* y2   = regA;                     // [4096][3520]  f32 (reuses y1)
  float* out1 = regA;                     // [4096][64]    f32 (after y2 dead)
  float* out2 = regA + 1048576;           // [2048][128]   f32

  long h = 0;
  _Float16* lin1_wh  = regH + h; h += 56623104;   // [64][512][1728]
  _Float16* w2h      = regH + h; h += 2097152;    // [64][64][512]
  _Float16* cl1_wh   = regH + h; h += 163840;     // [64][2560]
  _Float16* cl2_wh   = regH + h; h += 40960;      // [128][320]
  _Float16* fc1_wh   = regH + h; h += 1048576;    // [512][2048]
  _Float16* fc2_wh   = regH + h; h += 1024;       // [2][512]
  _Float16* Lr0h     = regH + h; h += 4096;
  _Float16* Lr1h     = regH + h; h += 1024;
  _Float16* o1h      = regH + h; h += 16252928;   // [c][b][32][124]
  _Float16* o2h      = regH + h; h += 7077888;    // [(c*64+b)][1728]
  _Float16* couth    = regH + h; h += 2097152;    // [(c*64+b)][512]
  _Float16* Xk1h     = regH + h; h += 10485760;   // [4096][2560]
  _Float16* Xk2h     = regH + h; h += 655360;     // [2048][320]
  _Float16* pooled2h = regH + h; h += 131072;     // [64][2048]
  _Float16* hh       = regH + h; h += 32768;      // [64][512]

  float* regS = (float*)(regH + h);
  float* m1 = regS;        float* v1 = m1 + 2048;
  float* m2 = v1 + 2048;   float* v2 = m2 + 4096;
  float* gm1 = v2 + 4096;  float* gv1 = gm1 + 64;
  float* gm2 = gv1 + 64;   float* gv2 = gm2 + 32;

  // ---- one-shot weight conversions to f16 --------------------------------
  f32_to_f16_kernel<<<cdiv(56623104, 256), 256, 0, stream>>>(lin1_w, lin1_wh, 56623104);
  f32_to_f16_kernel<<<cdiv(2097152, 256), 256, 0, stream>>>(conv2_w, w2h, 2097152);
  f32_to_f16_kernel<<<cdiv(163840, 256), 256, 0, stream>>>(cl1_w, cl1_wh, 163840);
  f32_to_f16_kernel<<<cdiv(40960, 256), 256, 0, stream>>>(cl2_w, cl2_wh, 40960);
  f32_to_f16_kernel<<<cdiv(1048576, 256), 256, 0, stream>>>(fc1_w, fc1_wh, 1048576);
  f32_to_f16_kernel<<<cdiv(1024, 256), 256, 0, stream>>>(fc2_w, fc2_wh, 1024);
  build_lr_kernel<<<16, 256, 0, stream>>>(L0, L1, lmax0, lmax1, Lr0h, Lr1h);

  // ---- stage 1: per-channel conv branches --------------------------------
  conv1_kernel<<<dim3(cdiv(64 * 249, 256), 2048), 256, 0, stream>>>(inp, conv1_w, conv1_b, y1);
  group_stats_kernel<<<2048, 256, 0, stream>>>(y1, 64L * 249, 64 * 249, m1, v1);
  bn1_pool_kernel<<<cdiv(64L * 32 * 64 * 124, 256), 256, 0, stream>>>(y1, m1, v1, bn1_g, bn1_b, o1h);

  conv2_wmma_kernel<<<dim3(220, 1, 64), 128, 0, stream>>>(o1h, w2h, conv2_b, y2);
  group_stats_kernel<<<4096, 256, 0, stream>>>(y2, 3520, 3520, m2, v2);
  bn2_pool_kernel<<<cdiv(64L * 64 * 64 * 27, 256), 256, 0, stream>>>(y2, m2, v2, bn2_g, bn2_b, o2h);

  // lin1 (batched, LDS-staged): [64x1728]*[512x1728]^T + b, relu -> f16
  gemm_bt_lds_kernel<_Float16><<<dim3(8, 1, 64), 128, 0, stream>>>(
      o2h, 1728, 64L * 1728, lin1_wh, 1728, 512L * 1728, lin1_b, 512,
      couth, 512, 64L * 512, 64, 512, 1728, 1);

  // ---- graph stage 1 ------------------------------------------------------
  scatter_x0_kernel<<<cdiv(64L * 64 * 512, 256), 256, 0, stream>>>(couth, perm, Xk1h);
  cheby_step_kernel<<<dim3(32, 1, 64), 128, 0, stream>>>(Lr0h, 64, Xk1h, 2560, 5, 0, 0, 1, 1.0f,  0.0f);
  cheby_step_kernel<<<dim3(32, 1, 64), 128, 0, stream>>>(Lr0h, 64, Xk1h, 2560, 5, 1, 0, 2, 2.0f, -1.0f);
  cheby_step_kernel<<<dim3(32, 1, 64), 128, 0, stream>>>(Lr0h, 64, Xk1h, 2560, 5, 2, 1, 3, 2.0f, -1.0f);
  cheby_step_kernel<<<dim3(32, 1, 64), 128, 0, stream>>>(Lr0h, 64, Xk1h, 2560, 5, 3, 2, 4, 2.0f, -1.0f);

  // [4096x2560]*[64x2560]^T + cl1_b -> f32
  gemm_bt_lds_kernel<float><<<dim3(1, 64, 1), 128, 0, stream>>>(
      Xk1h, 2560, 0, cl1_wh, 2560, 0, cl1_b, 0, out1, 64, 0, 4096, 64, 2560, 0);
  node_stats_kernel<<<64, 256, 0, stream>>>(out1, 64, 64, 64, gm1, gv1);
  gbn_pool_scatter_kernel<<<cdiv(64L * 32 * 64, 256), 256, 0, stream>>>(
      out1, 64, 64, gm1, gv1, 64.0f, gbn1_g, gbn1_b, Xk2h, 320, 5);

  // ---- graph stage 2 (V=32, F=64) -----------------------------------------
  cheby_step_kernel<<<dim3(4, 1, 64), 128, 0, stream>>>(Lr1h, 32, Xk2h, 320, 5, 0, 0, 1, 1.0f,  0.0f);
  cheby_step_kernel<<<dim3(4, 1, 64), 128, 0, stream>>>(Lr1h, 32, Xk2h, 320, 5, 1, 0, 2, 2.0f, -1.0f);
  cheby_step_kernel<<<dim3(4, 1, 64), 128, 0, stream>>>(Lr1h, 32, Xk2h, 320, 5, 2, 1, 3, 2.0f, -1.0f);
  cheby_step_kernel<<<dim3(4, 1, 64), 128, 0, stream>>>(Lr1h, 32, Xk2h, 320, 5, 3, 2, 4, 2.0f, -1.0f);

  // [2048x320]*[128x320]^T + cl2_b -> f32
  gemm_bt_lds_kernel<float><<<dim3(2, 32, 1), 128, 0, stream>>>(
      Xk2h, 320, 0, cl2_wh, 320, 0, cl2_b, 0, out2, 128, 0, 2048, 128, 320, 0);
  node_stats_kernel<<<32, 256, 0, stream>>>(out2, 32, 128, 64, gm2, gv2);
  gbn_pool_scatter_kernel<<<cdiv(64L * 16 * 128, 256), 256, 0, stream>>>(
      out2, 32, 128, gm2, gv2, 128.0f, gbn2_g, gbn2_b, pooled2h, 128, 1);

  // ---- classifier ---------------------------------------------------------
  gemm_bt_lds_kernel<_Float16><<<dim3(8, 1, 1), 128, 0, stream>>>(
      pooled2h, 2048, 0, fc1_wh, 2048, 0, fc1_b, 0, hh, 512, 0, 64, 512, 2048, 1);
  gemm_bt_h_kernel<float><<<dim3(1, 1, 1), 128, 0, stream>>>(
      hh, 512, 0, fc2_wh, 512, 0, fc2_b, 0, out, 2, 0, 64, 2, 512, 0);
}
